// SimilarityMatrix_53944789238351
// MI455X (gfx1250) — compile-verified, matrix operand order checked by reading, not run on hardware
//
#include <hip/hip_runtime.h>
#include <hip/hip_bf16.h>

// Problem dims (fixed by the reference)
#define NUM_HEADS 4
#define HEAD_DIM  128
#define MODEL_DIM 512
#define INPUT_DIM 512
#define BATCH     16
#define FDIM      1024

typedef __attribute__((ext_vector_type(16))) __bf16 v16bf;
typedef __attribute__((ext_vector_type(8)))  __bf16 v8bf;
typedef __attribute__((ext_vector_type(8)))  float  v8f;

// ---------------------------------------------------------------------------
// Fragment loaders matching the CDNA5 WMMA 16-bit A/B VGPR layout
// (ISA 7.12.2): lane l holds row (l&15); VGPRs 0-3 hold K = (l>>4)*8 .. +7,
// VGPRs 4-7 hold K = 16+(l>>4)*8 .. +7.  Two 16B contiguous chunks per lane.
// ---------------------------------------------------------------------------
static __device__ __forceinline__ v16bf
load_frag_f32_as_bf16(const float* __restrict__ base, int stride, int lane) {
    const int r  = lane & 15;
    const int kb = (lane >> 4) << 3;           // 0 or 8
    const float* p = base + (size_t)r * stride + kb;
    float4 c0 = *(const float4*)(p);
    float4 c1 = *(const float4*)(p + 4);
    float4 c2 = *(const float4*)(p + 16);
    float4 c3 = *(const float4*)(p + 20);
    float tmp[16] = {c0.x, c0.y, c0.z, c0.w, c1.x, c1.y, c1.z, c1.w,
                     c2.x, c2.y, c2.z, c2.w, c3.x, c3.y, c3.z, c3.w};
    v16bf out;
#pragma unroll
    for (int i = 0; i < 16; ++i) out[i] = (__bf16)tmp[i];
    return out;
}

static __device__ __forceinline__ v16bf
load_frag_bf16(const __bf16* __restrict__ base, int stride, int lane) {
    const int r  = lane & 15;
    const int kb = (lane >> 4) << 3;           // 0 or 8
    const __bf16* p = base + (size_t)r * stride + kb;
    v8bf lo = *(const v8bf*)(p);               // 16B load
    v8bf hi = *(const v8bf*)(p + 16);          // 16B load
    v16bf out;
#pragma unroll
    for (int i = 0; i < 8; ++i) { out[i] = lo[i]; out[8 + i] = hi[i]; }
    return out;
}

// ---------------------------------------------------------------------------
// Kernel 1: Y[m,n] = sum_k X[m,k] * W[n,k] + bias[n], stored as bf16.
// Wave computes a 16(M) x 64(N) tile; 8 waves/block stacked along M.
// grid = (M/128, N/64)
// ---------------------------------------------------------------------------
__global__ void __launch_bounds__(256)
proj_bf16_kernel(const float* __restrict__ X,     // [M, INPUT_DIM]
                 const float* __restrict__ W,     // [MODEL_DIM, INPUT_DIM]
                 const float* __restrict__ bias,  // [MODEL_DIM]
                 __bf16* __restrict__ Y)          // [M, MODEL_DIM] bf16
{
    const int lane = threadIdx.x & 31;
    const int wid  = threadIdx.x >> 5;
    const int M0   = blockIdx.x * 128 + wid * 16;
    const int N0   = blockIdx.y * 64;

    v8f acc[4] = {};
    const float* Abase = X + (size_t)M0 * INPUT_DIM;

#pragma unroll 1
    for (int k = 0; k < INPUT_DIM; k += 32) {
        v16bf a = load_frag_f32_as_bf16(Abase + k, INPUT_DIM, lane);
#pragma unroll
        for (int t = 0; t < 4; ++t) {
            v16bf b = load_frag_f32_as_bf16(
                W + (size_t)(N0 + t * 16) * INPUT_DIM + k, INPUT_DIM, lane);
            acc[t] = __builtin_amdgcn_wmma_f32_16x16x32_bf16(
                false, a, false, b, (short)0, acc[t], false, false);
        }
    }

    // C/D layout: VGPR r -> row r + 8*(lane>>4); col = (lane&15) + 16*t + N0
    const int colLo = N0 + (lane & 15);
    const int rowHi = (lane >> 4) * 8;
#pragma unroll
    for (int t = 0; t < 4; ++t) {
        const int col = colLo + t * 16;
        const float bv = bias[col];
#pragma unroll
        for (int r = 0; r < 8; ++r) {
            const int row = M0 + rowHi + r;
            Y[(size_t)row * MODEL_DIM + col] = (__bf16)(acc[t][r] + bv);
        }
    }
}

// ---------------------------------------------------------------------------
// Kernel 2: scores = softmax( (Qh Kh^T) / sqrt(128) ), one block per
// (b, h, 16-row tile). 8 waves; wave w owns columns [w*128, w*128+128).
// All 16x1024 scores stay in registers (8 accumulators per wave).
// ---------------------------------------------------------------------------
__global__ void __launch_bounds__(256)
attn_softmax_kernel(const __bf16* __restrict__ Q,  // [B, F, MODEL_DIM] bf16
                    const __bf16* __restrict__ K,  // [B, F, MODEL_DIM] bf16
                    float* __restrict__ out)       // [B*H, F, F] f32
{
    __shared__ float red[16][8];

    const int lane = threadIdx.x & 31;
    const int w    = threadIdx.x >> 5;
    const int bh   = blockIdx.x >> 6;      // b*NUM_HEADS + h
    const int rt   = blockIdx.x & 63;      // 16-row tile index
    const int b    = bh >> 2;
    const int h    = bh & 3;
    const int row0 = rt * 16;

    const __bf16* Qbase = Q + ((size_t)b * FDIM + row0) * MODEL_DIM + h * HEAD_DIM;
    const __bf16* Kbase = K + (size_t)b * FDIM * MODEL_DIM + h * HEAD_DIM;

    v8f acc[8] = {};
#pragma unroll
    for (int k = 0; k < HEAD_DIM; k += 32) {
        v16bf a = load_frag_bf16(Qbase + k, MODEL_DIM, lane);
#pragma unroll
        for (int t = 0; t < 8; ++t) {
            const int col = w * 128 + t * 16;
            v16bf bf = load_frag_bf16(Kbase + (size_t)col * MODEL_DIM + k,
                                      MODEL_DIM, lane);
            acc[t] = __builtin_amdgcn_wmma_f32_16x16x32_bf16(
                false, a, false, bf, (short)0, acc[t], false, false);
        }
    }

    const float scale = 0.08838834764831845f;  // 1/sqrt(HEAD_DIM)
    const int rowHi = (lane >> 4) * 8;

    // ---- row-wise max (raw scores; scale is positive so max commutes) ----
    float rmax[8];
#pragma unroll
    for (int r = 0; r < 8; ++r) {
        float m = acc[0][r];
#pragma unroll
        for (int t = 1; t < 8; ++t) m = fmaxf(m, acc[t][r]);
        // reduce across the 16 lanes holding the same row (wave32)
#pragma unroll
        for (int mask = 1; mask < 16; mask <<= 1)
            m = fmaxf(m, __shfl_xor(m, mask, 32));
        rmax[r] = m;
    }
    if ((lane & 15) == 0) {
#pragma unroll
        for (int r = 0; r < 8; ++r) red[rowHi + r][w] = rmax[r];
    }
    __syncthreads();
#pragma unroll
    for (int r = 0; r < 8; ++r) {
        float m = red[rowHi + r][0];
#pragma unroll
        for (int ww = 1; ww < 8; ++ww) m = fmaxf(m, red[rowHi + r][ww]);
        rmax[r] = m;
    }
    __syncthreads();

    // ---- exp((s - max) * scale) and row-wise sum ----
    float rsum[8];
#pragma unroll
    for (int r = 0; r < 8; ++r) rsum[r] = 0.0f;
#pragma unroll
    for (int t = 0; t < 8; ++t) {
#pragma unroll
        for (int r = 0; r < 8; ++r) {
            float e = __expf((acc[t][r] - rmax[r]) * scale);
            acc[t][r] = e;
            rsum[r] += e;
        }
    }
#pragma unroll
    for (int r = 0; r < 8; ++r) {
        float s = rsum[r];
#pragma unroll
        for (int mask = 1; mask < 16; mask <<= 1)
            s += __shfl_xor(s, mask, 32);
        rsum[r] = s;
    }
    if ((lane & 15) == 0) {
#pragma unroll
        for (int r = 0; r < 8; ++r) red[rowHi + r][w] = rsum[r];
    }
    __syncthreads();
    float rinv[8];
#pragma unroll
    for (int r = 0; r < 8; ++r) {
        float s = 0.0f;
#pragma unroll
        for (int ww = 0; ww < 8; ++ww) s += red[rowHi + r][ww];
        rinv[r] = 1.0f / s;
    }

    // ---- normalize and store (lanes 0-15 contiguous columns per tile) ----
    float* obase = out + ((size_t)bh * FDIM + row0) * FDIM;
    const int colLo = w * 128 + (lane & 15);
#pragma unroll
    for (int t = 0; t < 8; ++t) {
        const int col = colLo + t * 16;
#pragma unroll
        for (int r = 0; r < 8; ++r)
            obase[(size_t)(rowHi + r) * FDIM + col] = acc[t][r] * rinv[r];
    }
}

// ---------------------------------------------------------------------------
extern "C" void kernel_launch(void* const* d_in, const int* in_sizes, int n_in,
                              void* d_out, int out_size, void* d_ws, size_t ws_size,
                              hipStream_t stream) {
    (void)in_sizes; (void)n_in; (void)out_size; (void)ws_size;
    const float* query = (const float*)d_in[0];
    const float* key   = (const float*)d_in[1];
    // d_in[2] = value : projected result is discarded by the reference -> skip
    const float* Wq = (const float*)d_in[3];
    const float* bq = (const float*)d_in[4];
    const float* Wk = (const float*)d_in[5];
    const float* bk = (const float*)d_in[6];
    // d_in[7] = Wv, d_in[8] = bv : unused

    __bf16* wsQ = (__bf16*)d_ws;                               // 16 MB
    __bf16* wsK = wsQ + (size_t)BATCH * FDIM * MODEL_DIM;      // +16 MB

    const int M = BATCH * FDIM;                 // 16384
    dim3 pgrid(M / 128, MODEL_DIM / 64);        // (128, 8)
    proj_bf16_kernel<<<pgrid, 256, 0, stream>>>(query, Wq, bq, wsQ);
    proj_bf16_kernel<<<pgrid, 256, 0, stream>>>(key,   Wk, bk, wsK);

    const int nblocks = BATCH * NUM_HEADS * (FDIM / 16);       // 4096
    attn_softmax_kernel<<<nblocks, 256, 0, stream>>>(wsQ, wsK, (float*)d_out);
}
